// RotaryAttention_42417097015902
// MI455X (gfx1250) — matrix operand order repaired and checked by reading, compiled, not run on hardware
//
#include <hip/hip_runtime.h>
#include <hip/hip_bf16.h>

#define D_MODEL 1024
#define N_HEADS 16
#define HEAD_DIM 64
#define SEQ 2048
#define BATCH 2
#define M_ROWS (SEQ * BATCH)   // 4096

typedef __attribute__((ext_vector_type(16))) _Float16 v16h;
typedef __attribute__((ext_vector_type(8)))  _Float16 v8h;
typedef __attribute__((ext_vector_type(8)))  float    v8f;
typedef __attribute__((ext_vector_type(4)))  unsigned int u32x4;
typedef __attribute__((ext_vector_type(8)))  int          i32x8;
typedef __attribute__((ext_vector_type(4)))  int          i32x4;

union V16 { v16h v; v8h h[2]; _Float16 e[16]; };
union V8  { v8h  v; _Float16 e[8]; };

#if defined(__gfx1250__) && __has_builtin(__builtin_amdgcn_tensor_load_to_lds) && \
    __has_builtin(__builtin_amdgcn_s_wait_tensorcnt)
#define USE_TDM 1
#endif

__device__ __forceinline__ v8f wmma_f16(v16h a, v16h b, v8f c) {
  // D = A(16x32 f16) * B(32x16 f16) + C(16x16 f32)
  return __builtin_amdgcn_wmma_f32_16x16x32_f16(
      false, a, false, b, (short)0, c, false, false);
}

// A fragment: lane holds one M-row; elements j -> K = k + hi*16 + j
__device__ __forceinline__ v16h loadA(const _Float16* __restrict__ row, int k, int hi) {
  V16 a;
  a.h[0] = *(const v8h*)(row + k + hi * 16);
  a.h[1] = *(const v8h*)(row + k + hi * 16 + 8);
  return a.v;
}
// B fragment: lane holds one N-col (= row of W); elements j -> K = k + hi*16 + j
__device__ __forceinline__ v16h loadB(const _Float16* __restrict__ row, int k, int hi) {
  V16 b;
  b.h[0] = *(const v8h*)(row + k + hi * 16);
  b.h[1] = *(const v8h*)(row + k + hi * 16 + 8);
  return b.v;
}

// ---------------------------------------------------------------------------
// Kernel 0: bulk f32 -> f16 conversion (activations + weights), one pass.
// ---------------------------------------------------------------------------
__global__ __launch_bounds__(256) void cvt_f16_kernel(
    const float* __restrict__ src, _Float16* __restrict__ dst, int n8)
{
  const int i = blockIdx.x * 256 + threadIdx.x;
  if (i >= n8) return;
  const float4* p = (const float4*)src + (size_t)i * 2;
  const float4 x = p[0], y = p[1];
  V8 o;
  o.e[0] = (_Float16)x.x; o.e[1] = (_Float16)x.y;
  o.e[2] = (_Float16)x.z; o.e[3] = (_Float16)x.w;
  o.e[4] = (_Float16)y.x; o.e[5] = (_Float16)y.y;
  o.e[6] = (_Float16)y.z; o.e[7] = (_Float16)y.w;
  *(v8h*)(dst + (size_t)i * 8) = o.v;
}

// ---------------------------------------------------------------------------
// Kernel 1: Y = X @ W^T + b (all-f16 operands, f32 accum), optional RoPE,
// output (BATCH, N_HEADS, SEQ, HEAD_DIM) f16. Wave = 16 rows x 64 cols.
// ---------------------------------------------------------------------------
__global__ __launch_bounds__(256) void qkv_rope_kernel(
    const _Float16* __restrict__ X,  // (M_ROWS, D_MODEL) f16, row = s*BATCH+b
    const _Float16* __restrict__ W,  // (D_MODEL, D_MODEL) f16
    const float* __restrict__ bias,  // (D_MODEL) f32
    _Float16* __restrict__ out,      // (BATCH, N_HEADS, SEQ, HEAD_DIM)
    int apply_rope, float scale)
{
  const int tid  = threadIdx.x;
  const int lane = tid & 31;
  const int wave = tid >> 5;
  const int hi   = (lane >> 4) & 1;
  const int ln   = lane & 15;
  const int head = blockIdx.y;
  const int mbase = blockIdx.x * 128 + wave * 16;

  v8f acc[4];
#pragma unroll
  for (int t = 0; t < 4; ++t)
#pragma unroll
    for (int r = 0; r < 8; ++r) acc[t][r] = 0.0f;

  const _Float16* xrow = X + (size_t)(mbase + ln) * D_MODEL;
  const _Float16* wrow[4];
#pragma unroll
  for (int dt = 0; dt < 4; ++dt)
    wrow[dt] = W + (size_t)(head * HEAD_DIM + dt * 16 + ln) * D_MODEL;

  for (int k = 0; k < D_MODEL; k += 32) {
    const v16h a = loadA(xrow, k, hi);
    if (k + 32 < D_MODEL) __builtin_prefetch(xrow + k + 32, 0, 0);
#pragma unroll
    for (int dt = 0; dt < 4; ++dt)
      acc[dt] = wmma_f16(a, loadB(wrow[dt], k, hi), acc[dt]);
  }

#pragma unroll
  for (int dt = 0; dt < 4; ++dt) {
    const float bv = bias[head * HEAD_DIM + dt * 16 + ln];
#pragma unroll
    for (int r = 0; r < 8; ++r) acc[dt][r] += bv;
  }

  if (apply_rope) {
    // pair (d, d+32) lives in (acc[dt], acc[dt+2]) of the same lane
#pragma unroll
    for (int dt = 0; dt < 2; ++dt) {
      const int i = dt * 16 + ln;                       // rotary index 0..31
      const float theta = __powf(10000.0f, -((float)i) / 32.0f);
#pragma unroll
      for (int r = 0; r < 8; ++r) {
        const int mrow = mbase + r + hi * 8;
        const int s = mrow >> 1;                        // row = s*BATCH + b
        const float ang = (float)s * theta;
        const float sn = __sinf(ang), cs = __cosf(ang);
        const float lo = acc[dt][r];
        const float hp = acc[dt + 2][r];
        acc[dt][r]     = lo * cs - hp * sn;             // reference rope
        acc[dt + 2][r] = lo * sn - hp * cs;             // (q1*sin - q2*cos)
      }
    }
  }

#pragma unroll
  for (int dt = 0; dt < 4; ++dt) {
    const int d = dt * 16 + ln;
#pragma unroll
    for (int r = 0; r < 8; ++r) {
      const int mrow = mbase + r + hi * 8;
      const int s = mrow >> 1, b = mrow & 1;
      out[(((size_t)b * N_HEADS + head) * SEQ + s) * HEAD_DIM + d] =
          (_Float16)(acc[dt][r] * scale);
    }
  }
}

// ---------------------------------------------------------------------------
// Kernel 2: flash attention; one (b,h) per block-y, 128 queries per block
// (8 waves x 16 queries). K/V chunks of 32 keys staged in LDS via the
// Tensor Data Mover when available (descriptor-driven DMA + s_wait_tensorcnt).
// ---------------------------------------------------------------------------
__global__ __launch_bounds__(256) void attention_kernel(
    const _Float16* __restrict__ qws,   // (B,H,S,64), pre-scaled by 1/8
    const _Float16* __restrict__ kws,   // (B,H,S,64)
    const _Float16* __restrict__ vws,   // (B,H,S,64)
    _Float16* __restrict__ aws)         // (S,B,D_MODEL)
{
  __shared__ _Float16 Klds[32 * 64];        // [key][d]
  __shared__ _Float16 Vt[64 * 32];          // [d][key] (transposed)
  __shared__ _Float16 Plds[8][16 * 32];     // per-wave P tile, [m][key]
#ifdef USE_TDM
  __shared__ _Float16 Vrow[32 * 64];        // TDM lands V row-major here
#endif

  const int tid  = threadIdx.x;
  const int lane = tid & 31;
  const int wave = tid >> 5;
  const int hi   = (lane >> 4) & 1;
  const int ln   = lane & 15;
  const int bh   = blockIdx.y;
  const int b    = bh >> 4, h = bh & 15;
  const int qbase = blockIdx.x * 128 + wave * 16;

  const _Float16* qptr = qws + (size_t)bh * SEQ * HEAD_DIM;
  const _Float16* kptr = kws + (size_t)bh * SEQ * HEAD_DIM;
  const _Float16* vptr = vws + (size_t)bh * SEQ * HEAD_DIM;

  // Q A-fragments for d=[0,32) and d=[32,64)
  V16 aq[2];
  {
    const _Float16* pq = qptr + (size_t)(qbase + ln) * HEAD_DIM;
#pragma unroll
    for (int c = 0; c < 2; ++c) {
      aq[c].h[0] = *(const v8h*)(pq + c * 32 + hi * 16);
      aq[c].h[1] = *(const v8h*)(pq + c * 32 + hi * 16 + 8);
    }
  }

  float m_i[8], l_i[8];
  v8f acco[4];
#pragma unroll
  for (int r = 0; r < 8; ++r) { m_i[r] = -1e30f; l_i[r] = 0.0f; }
#pragma unroll
  for (int dt = 0; dt < 4; ++dt)
#pragma unroll
    for (int r = 0; r < 8; ++r) acco[dt][r] = 0.0f;

  for (int k0 = 0; k0 < SEQ; k0 += 32) {
    __syncthreads();   // all waves done reading previous tiles

#ifdef USE_TDM
    if (wave == 0) {
      // ---- Tensor DMA descriptors: 2D tile 32(keys) x 64(d), f16 ----
      const unsigned long long gK =
          (unsigned long long)(uintptr_t)(kptr + (size_t)k0 * HEAD_DIM);
      const unsigned long long gV =
          (unsigned long long)(uintptr_t)(vptr + (size_t)k0 * HEAD_DIM);
      const unsigned int ldsK = (unsigned int)(uintptr_t)&Klds[0];
      const unsigned int ldsV = (unsigned int)(uintptr_t)&Vrow[0];
      // group0: count=1 | lds_addr | global_addr[56:0] | type=2
      u32x4 g0K = { 1u, ldsK, (unsigned int)gK,
                    (unsigned int)(gK >> 32) | (2u << 30) };
      u32x4 g0V = { 1u, ldsV, (unsigned int)gV,
                    (unsigned int)(gV >> 32) | (2u << 30) };
      // group1: data_size=2B; tensor_dim0=64, tensor_dim1=SEQ;
      //         tile_dim0=64, tile_dim1=32; tensor_dim0_stride=64
      i32x8 g1 = { (int)(1u << 16),
                   (int)((unsigned)HEAD_DIM << 16),
                   (int)(((unsigned)SEQ & 0xFFFFu) << 16),
                   (int)(((unsigned)SEQ >> 16) | ((unsigned)HEAD_DIM << 16)),
                   32,
                   HEAD_DIM,
                   0, 0 };
      i32x4 z4 = { 0, 0, 0, 0 };
      i32x8 z8 = { 0, 0, 0, 0, 0, 0, 0, 0 };
      __builtin_amdgcn_tensor_load_to_lds(g0K, g1, z4, z4, z8, 0);
      __builtin_amdgcn_tensor_load_to_lds(g0V, g1, z4, z4, z8, 0);
      __builtin_amdgcn_s_wait_tensorcnt((short)0);
    }
    __syncthreads();   // DMA'd tiles visible to all waves
    {
      // transpose V within LDS: [key][d] -> [d][key]
      const int row = tid >> 3, c8 = (tid & 7) * 8;
      V8 vv; vv.v = *(const v8h*)&Vrow[row * 64 + c8];
#pragma unroll
      for (int i = 0; i < 8; ++i) Vt[(c8 + i) * 32 + row] = vv.e[i];
    }
    __syncthreads();
#else
    {
      // fallback: cooperative global->reg->LDS staging
      const int row = tid >> 3, c8 = (tid & 7) * 8;
      *(v8h*)&Klds[row * 64 + c8] =
          *(const v8h*)(kptr + (size_t)(k0 + row) * HEAD_DIM + c8);
      V8 vv;
      vv.v = *(const v8h*)(vptr + (size_t)(k0 + row) * HEAD_DIM + c8);
#pragma unroll
      for (int i = 0; i < 8; ++i) Vt[(c8 + i) * 32 + row] = vv.e[i];
      if (k0 + 32 < SEQ) {
        __builtin_prefetch(kptr + (size_t)(k0 + 32 + row) * HEAD_DIM + c8, 0, 0);
        __builtin_prefetch(vptr + (size_t)(k0 + 32 + row) * HEAD_DIM + c8, 0, 0);
      }
    }
    __syncthreads();
#endif

    // scores S = Q @ K^T for 32 keys: 2 key-subtiles x 2 d-chunks
    v8f sc[2];
#pragma unroll
    for (int t = 0; t < 2; ++t)
#pragma unroll
      for (int r = 0; r < 8; ++r) sc[t][r] = 0.0f;
#pragma unroll
    for (int t = 0; t < 2; ++t) {
      const int key = t * 16 + ln;
#pragma unroll
      for (int c = 0; c < 2; ++c) {
        V16 bk;
        const _Float16* pk = &Klds[key * 64 + c * 32 + hi * 16];
        bk.h[0] = *(const v8h*)pk;
        bk.h[1] = *(const v8h*)(pk + 8);
        sc[t] = wmma_f16(aq[c].v, bk.v, sc[t]);
      }
    }

    // online softmax (rows live in (reg, lane-half); keys across 16 lanes)
    float alpha[8];
#pragma unroll
    for (int r = 0; r < 8; ++r) {
      float v = fmaxf(sc[0][r], sc[1][r]);
#pragma unroll
      for (int off = 1; off < 16; off <<= 1)
        v = fmaxf(v, __shfl_xor(v, off, 32));
      const float mn = fmaxf(m_i[r], v);
      alpha[r] = __expf(m_i[r] - mn);
      m_i[r] = mn;
      const float p0 = __expf(sc[0][r] - mn);
      const float p1 = __expf(sc[1][r] - mn);
      sc[0][r] = p0; sc[1][r] = p1;
      float rs = p0 + p1;
#pragma unroll
      for (int off = 1; off < 16; off <<= 1)
        rs += __shfl_xor(rs, off, 32);
      l_i[r] = l_i[r] * alpha[r] + rs;
    }

    // transpose P through LDS into A-fragment layout
#pragma unroll
    for (int t = 0; t < 2; ++t)
#pragma unroll
      for (int r = 0; r < 8; ++r)
        Plds[wave][(r + hi * 8) * 32 + t * 16 + ln] = (_Float16)sc[t][r];

#pragma unroll
    for (int dt = 0; dt < 4; ++dt)
#pragma unroll
      for (int r = 0; r < 8; ++r) acco[dt][r] *= alpha[r];

    V16 ap;
    {
      const _Float16* pp = &Plds[wave][ln * 32];
      ap.h[0] = *(const v8h*)(pp + hi * 16);
      ap.h[1] = *(const v8h*)(pp + hi * 16 + 8);
    }

    // out += P(16x32) @ V(32x64)
#pragma unroll
    for (int dt = 0; dt < 4; ++dt) {
      V16 bv;
      const _Float16* pv = &Vt[(dt * 16 + ln) * 32 + hi * 16];
      bv.h[0] = *(const v8h*)pv;
      bv.h[1] = *(const v8h*)(pv + 8);
      acco[dt] = wmma_f16(ap.v, bv.v, acco[dt]);
    }
  }

  // normalize and store to (S,B,D_MODEL) f16
#pragma unroll
  for (int dt = 0; dt < 4; ++dt) {
    const int d = h * HEAD_DIM + dt * 16 + ln;
#pragma unroll
    for (int r = 0; r < 8; ++r) {
      const int q = qbase + r + hi * 8;
      aws[((size_t)q * BATCH + b) * D_MODEL + d] =
          (_Float16)(acco[dt][r] / l_i[r]);
    }
  }
}

// ---------------------------------------------------------------------------
// Kernel 3: out = attn @ Wo^T + bo (f16 operands, f32 out)
// ---------------------------------------------------------------------------
__global__ __launch_bounds__(256) void out_proj_kernel(
    const _Float16* __restrict__ A,   // (M_ROWS, D_MODEL) f16
    const _Float16* __restrict__ W,   // (D_MODEL, D_MODEL) f16
    const float* __restrict__ bo,     // (D_MODEL)
    float* __restrict__ out)          // (M_ROWS, D_MODEL)
{
  const int tid  = threadIdx.x;
  const int lane = tid & 31;
  const int wave = tid >> 5;
  const int hi   = (lane >> 4) & 1;
  const int ln   = lane & 15;
  const int mbase = blockIdx.x * 128 + wave * 16;
  const int ntile = blockIdx.y;

  v8f acc[4];
#pragma unroll
  for (int t = 0; t < 4; ++t)
#pragma unroll
    for (int r = 0; r < 8; ++r) acc[t][r] = 0.0f;

  const _Float16* pa = A + (size_t)(mbase + ln) * D_MODEL;
  const _Float16* wrow[4];
#pragma unroll
  for (int dt = 0; dt < 4; ++dt)
    wrow[dt] = W + (size_t)(ntile * 64 + dt * 16 + ln) * D_MODEL;

  for (int k = 0; k < D_MODEL; k += 32) {
    const v16h a = loadA(pa, k, hi);
    if (k + 32 < D_MODEL) __builtin_prefetch(pa + k + 32, 0, 0);
#pragma unroll
    for (int dt = 0; dt < 4; ++dt)
      acc[dt] = wmma_f16(a, loadB(wrow[dt], k, hi), acc[dt]);
  }

#pragma unroll
  for (int dt = 0; dt < 4; ++dt) {
    const int n = ntile * 64 + dt * 16 + ln;
    const float bv = bo[n];
#pragma unroll
    for (int r = 0; r < 8; ++r) {
      const int mrow = mbase + r + hi * 8;
      out[(size_t)mrow * D_MODEL + n] = acc[dt][r] + bv;
    }
  }
}

// ---------------------------------------------------------------------------
extern "C" void kernel_launch(void* const* d_in, const int* in_sizes, int n_in,
                              void* d_out, int out_size, void* d_ws, size_t ws_size,
                              hipStream_t stream) {
  const float* query = (const float*)d_in[0];
  const float* key_  = (const float*)d_in[1];
  const float* value = (const float*)d_in[2];
  const float* Wq = (const float*)d_in[3];
  const float* bq = (const float*)d_in[4];
  const float* Wk = (const float*)d_in[5];
  const float* bk = (const float*)d_in[6];
  const float* Wv = (const float*)d_in[7];
  const float* bv = (const float*)d_in[8];
  const float* Wo = (const float*)d_in[9];
  const float* bo = (const float*)d_in[10];

  const size_t ACT = (size_t)M_ROWS * D_MODEL;      // 4M elements
  const size_t WSZ = (size_t)D_MODEL * D_MODEL;     // 1M elements

  _Float16* p = (_Float16*)d_ws;
  _Float16* qh = p;  p += ACT;   // f16 copies of activations
  _Float16* kh = p;  p += ACT;
  _Float16* vh = p;  p += ACT;
  _Float16* wqh = p; p += WSZ;   // f16 copies of weights
  _Float16* wkh = p; p += WSZ;
  _Float16* wvh = p; p += WSZ;
  _Float16* woh = p; p += WSZ;
  _Float16* qws = p; p += ACT;   // projected+rope'd Q/K/V, (B,H,S,Dh)
  _Float16* kws = p; p += ACT;
  _Float16* vws = p; p += ACT;
  _Float16* aws = p; p += ACT;   // attention out, (S,B,D)

  const dim3 blk(256);

  // ---- bulk f32 -> f16 ----
  cvt_f16_kernel<<<dim3((int)(ACT / 8 / 256)), blk, 0, stream>>>(query, qh, (int)(ACT / 8));
  cvt_f16_kernel<<<dim3((int)(ACT / 8 / 256)), blk, 0, stream>>>(key_,  kh, (int)(ACT / 8));
  cvt_f16_kernel<<<dim3((int)(ACT / 8 / 256)), blk, 0, stream>>>(value, vh, (int)(ACT / 8));
  cvt_f16_kernel<<<dim3((int)(WSZ / 8 / 256)), blk, 0, stream>>>(Wq, wqh, (int)(WSZ / 8));
  cvt_f16_kernel<<<dim3((int)(WSZ / 8 / 256)), blk, 0, stream>>>(Wk, wkh, (int)(WSZ / 8));
  cvt_f16_kernel<<<dim3((int)(WSZ / 8 / 256)), blk, 0, stream>>>(Wv, wvh, (int)(WSZ / 8));
  cvt_f16_kernel<<<dim3((int)(WSZ / 8 / 256)), blk, 0, stream>>>(Wo, woh, (int)(WSZ / 8));

  // ---- projections + RoPE ----
  const dim3 g1(M_ROWS / 128, N_HEADS);
  qkv_rope_kernel<<<g1, blk, 0, stream>>>(qh, wqh, bq, qws, 1, 0.125f);
  qkv_rope_kernel<<<g1, blk, 0, stream>>>(kh, wkh, bk, kws, 1, 1.0f);
  qkv_rope_kernel<<<g1, blk, 0, stream>>>(vh, wvh, bv, vws, 0, 1.0f);

  // ---- attention ----
  attention_kernel<<<dim3(SEQ / 128, BATCH * N_HEADS), blk, 0, stream>>>(
      qws, kws, vws, aws);

  // ---- output projection ----
  out_proj_kernel<<<dim3(M_ROWS / 128, D_MODEL / 64), blk, 0, stream>>>(
      aws, woh, bo, (float*)d_out);
}
